// Encoder_54889682043573
// MI455X (gfx1250) — matrix operand order repaired and checked by reading, not compiled, not run on hardware
//
#include <hip/hip_runtime.h>
#include <hip/hip_bf16.h>
#include <hip/hip_fp16.h>

typedef __attribute__((ext_vector_type(16))) _Float16 v16h;
typedef __attribute__((ext_vector_type(8)))  _Float16 v8h;
typedef __attribute__((ext_vector_type(8)))  float    v8f;

#define H    256
#define BSZ  64
#define TSEQ 1024
#define DIN  64
#define G4   1024   // 4*H gate columns per direction

// ---------------- workspace layout (units: _Float16) ----------------
#define WS_XT     0                            // [T][B][DIN]    fp16
#define WS_WIH0   (WS_XT    + TSEQ*BSZ*DIN)    // [2][G4][DIN]
#define WS_WHH0   (WS_WIH0  + 2*G4*DIN)        // [2][G4][H]
#define WS_WIH12  (WS_WHH0  + 2*G4*H)          // [4][G4][2H]
#define WS_WHH12  (WS_WIH12 + 4*G4*2*H)        // [4][G4][H]
#define WS_YS     (WS_WHH12 + 4*G4*H)          // [T][B][H]      fp16

// ---------------- WMMA fragment loaders (16x16x32 f16) ----------------
// A (16x32, M x K), row-major source with leading dim ldk:
//   lanes 0-15: M=lane, K = k0..k0+15
//   lanes 16-31: M=lane-16, K = k0+16..k0+31
__device__ __forceinline__ v16h loadA(const _Float16* base, int ldk, int k0) {
  const int lane = threadIdx.x & 31;
  const int m    = lane & 15;
  const int kb   = (lane & 16);
  const _Float16* p = base + m * ldk + k0 + kb;
  v8h lo = *(const v8h*)p;
  v8h hi = *(const v8h*)(p + 8);
  v16h r;
#pragma unroll
  for (int i = 0; i < 8; ++i) { r[i] = lo[i]; r[i + 8] = hi[i]; }
  return r;
}

// B (32x16, K x N) where B[k][n] = W[n][k], W row-major (N,K) with leading dim ldk.
//   lanes 0-15:  N=lane,    K = k0+0..k0+15   (one contiguous 32B row read)
//   lanes 16-31: N=lane-16, K = k0+16..k0+31
__device__ __forceinline__ v16h loadB(const _Float16* wrow, int ldk, int k0) {
  const int lane = threadIdx.x & 31;
  const int n    = lane & 15;
  const int kb   = (lane & 16);
  return *(const v16h*)(wrow + (size_t)n * ldk + k0 + kb);
}

// ---------------- prep kernels ----------------
__global__ void f32_to_f16(const float* __restrict__ src, _Float16* __restrict__ dst, int n) {
  int i = blockIdx.x * blockDim.x + threadIdx.x;
  if (i < n) dst[i] = (_Float16)src[i];
}

__global__ void transpose_x(const float* __restrict__ x, _Float16* __restrict__ xT) {
  // x: [B][DIN][T] fp32  ->  xT: [T][B][DIN] fp16
  int i = blockIdx.x * blockDim.x + threadIdx.x;
  if (i >= TSEQ * BSZ * DIN) return;
  int k = i & (DIN - 1);
  int b = (i >> 6) & (BSZ - 1);
  int t = i >> 12;
  xT[i] = (_Float16)x[((size_t)b * DIN + k) * TSEQ + t];
}

// ---------------- recurrent LSTM kernel ----------------
// grid = 4 blocks (batch slices of 16), block = 512 threads = 16 waves.
// Each wave owns 4 N-tiles (64 gate columns of 1024).
__global__ void __launch_bounds__(512, 1)
lstm_recurrent(const _Float16* __restrict__ xT,
               const _Float16* __restrict__ Wih0,
               const _Float16* __restrict__ Whh0,
               const _Float16* __restrict__ Wih12,
               const _Float16* __restrict__ Whh12,
               const float*    __restrict__ b0v,
               const float*    __restrict__ b12v,
               _Float16*       __restrict__ ys) {
  extern __shared__ char smem[];
  _Float16* hS = (_Float16*)smem;                        // [6][16][H] fp16   48KB
  float*    cS = (float*)(smem + 6 * 16 * H * 2);        // [6][16][H] f32    96KB
  float*    gS = (float*)((char*)cS + 6 * 16 * H * 4);   // [16][G4]  f32     64KB
  _Float16* xS = (_Float16*)((char*)gS + 16 * G4 * 4);   // [16][DIN] fp16     2KB

  const int tid   = threadIdx.x;
  const int wv    = tid >> 5;
  const int lane  = tid & 31;
  const int bbase = blockIdx.x * 16;
  const int n0    = wv * 64;            // first gate column of this wave

  for (int i = tid; i < 6 * 16 * H; i += 512) { hS[i] = (_Float16)0.f; cS[i] = 0.f; }
  __syncthreads();

  for (int t = 0; t < TSEQ; ++t) {
    // stage x_t tile for this batch slice: 16 x 64 halves = 512 dwords
    {
      const uint32_t* src = (const uint32_t*)(xT + (size_t)(t * BSZ + bbase) * DIN);
      ((uint32_t*)xS)[tid] = src[tid];
      if (t + 1 < TSEQ)
        __builtin_prefetch(xT + (size_t)((t + 1) * BSZ + bbase) * DIN, 0, 0);
    }
    __syncthreads();

    for (int l = 0; l < 3; ++l) {
      for (int d = 0; d < 2; ++d) {
        const int idx = 2 * l + d;
        v8f acc[4] = {};

        // ---- input-projection contribution ----
        if (l == 0) {
          const _Float16* W = Wih0 + (size_t)d * G4 * DIN;
#pragma unroll
          for (int k0 = 0; k0 < DIN; k0 += 32) {
            v16h a = loadA(xS, DIN, k0);
#pragma unroll
            for (int j = 0; j < 4; ++j) {
              v16h b = loadB(W + (size_t)(n0 + 16 * j) * DIN, DIN, k0);
              acc[j] = __builtin_amdgcn_wmma_f32_16x16x32_f16(
                  false, a, false, b, (short)0, acc[j], false, false);
            }
          }
        } else {
          const _Float16* W = Wih12 + (size_t)((l - 1) * 2 + d) * G4 * (2 * H);
#pragma unroll 4
          for (int k0 = 0; k0 < 2 * H; k0 += 32) {
            // input = concat(h[2(l-1)], h[2(l-1)+1]) along K
            const _Float16* hsrc = hS + (size_t)(2 * (l - 1) + (k0 >= H)) * 16 * H;
            v16h a = loadA(hsrc, H, k0 & (H - 1));
#pragma unroll
            for (int j = 0; j < 4; ++j) {
              v16h b = loadB(W + (size_t)(n0 + 16 * j) * (2 * H), 2 * H, k0);
              acc[j] = __builtin_amdgcn_wmma_f32_16x16x32_f16(
                  false, a, false, b, (short)0, acc[j], false, false);
            }
          }
        }

        // ---- recurrent contribution (reads OLD h[idx]) ----
        {
          const _Float16* Wh = (l == 0) ? (Whh0 + (size_t)d * G4 * H)
                                        : (Whh12 + (size_t)((l - 1) * 2 + d) * G4 * H);
          const _Float16* hsrc = hS + (size_t)idx * 16 * H;
#pragma unroll 4
          for (int k0 = 0; k0 < H; k0 += 32) {
            v16h a = loadA(hsrc, H, k0);
#pragma unroll
            for (int j = 0; j < 4; ++j) {
              v16h b = loadB(Wh + (size_t)(n0 + 16 * j) * H, H, k0);
              acc[j] = __builtin_amdgcn_wmma_f32_16x16x32_f16(
                  false, a, false, b, (short)0, acc[j], false, false);
            }
          }
        }

        // ---- spill gate pre-activations (C layout: VGPR g -> M = g + 8*(lane/16)) ----
        {
          const int n   = lane & 15;
          const int mhi = (lane >> 4) * 8;
#pragma unroll
          for (int j = 0; j < 4; ++j) {
            const int col = n0 + 16 * j + n;
#pragma unroll
            for (int g = 0; g < 8; ++g)
              gS[(size_t)(g + mhi) * G4 + col] = acc[j][g];
          }
        }
        __syncthreads();

        // ---- elementwise LSTM cell update ----
        {
          const float* bias = (l == 0) ? (b0v + (size_t)d * G4)
                                       : (b12v + (size_t)((l - 1) * 2 + d) * G4);
          for (int e = tid; e < 16 * H; e += 512) {
            const int m = e >> 8;
            const int j = e & (H - 1);
            const float* gr = gS + (size_t)m * G4;
            float ig = gr[j]         + bias[j];
            float fg = gr[H + j]     + bias[H + j];
            float gg = gr[2 * H + j] + bias[2 * H + j];
            float og = gr[3 * H + j] + bias[3 * H + j];
            float si = 1.f / (1.f + __expf(-ig));
            float sf = 1.f / (1.f + __expf(-fg));
            float so = 1.f / (1.f + __expf(-og));
            float eg = __expf(2.f * gg);
            float tg = (eg - 1.f) / (eg + 1.f);
            float cn = sf * cS[(size_t)idx * 16 * H + e] + si * tg;
            float ec = __expf(2.f * cn);
            float tc = (ec - 1.f) / (ec + 1.f);
            float hn = so * tc;
            cS[(size_t)idx * 16 * H + e] = cn;
            hS[(size_t)idx * 16 * H + e] = (_Float16)hn;
            if (idx == 5)
              ys[((size_t)t * BSZ + bbase + m) * H + j] = (_Float16)hn;
          }
        }
        __syncthreads();
      }
    }
  }
}

// ---------------- classifier: logits + argmax over 81 outputs ----------------
// softmax is monotone -> argmax of logits. One block per timestep, one thread per batch row.
__global__ void __launch_bounds__(64, 1)
classify(const _Float16* __restrict__ ys, const float* __restrict__ cw,
         const float* __restrict__ cb, int* __restrict__ out) {
  const int t = blockIdx.x;
  const int b = threadIdx.x;
  const _Float16* yr = ys + ((size_t)t * BSZ + b) * H;
  float best = -3.0e38f;
  int besto = 0;
  for (int o = 0; o < 81; ++o) {
    const float* w = cw + (size_t)o * H;
    float s = cb[o];
#pragma unroll 8
    for (int k = 0; k < H; ++k) s += w[k] * (float)yr[k];
    if (s > best) { best = s; besto = o; }
  }
  out[(size_t)b * TSEQ + t] = besto;   // (B, T)
}

// ---------------- launch ----------------
extern "C" void kernel_launch(void* const* d_in, const int* in_sizes, int n_in,
                              void* d_out, int out_size, void* d_ws, size_t ws_size,
                              hipStream_t stream) {
  const float* x      = (const float*)d_in[0];
  const float* Wih0f  = (const float*)d_in[1];
  const float* Whh0f  = (const float*)d_in[2];
  const float* b0v    = (const float*)d_in[3];
  const float* Wih12f = (const float*)d_in[4];
  const float* Whh12f = (const float*)d_in[5];
  const float* b12v   = (const float*)d_in[6];
  const float* convw  = (const float*)d_in[7];
  const float* convb  = (const float*)d_in[8];

  _Float16* ws      = (_Float16*)d_ws;
  _Float16* wXT     = ws + WS_XT;
  _Float16* wWih0   = ws + WS_WIH0;
  _Float16* wWhh0   = ws + WS_WHH0;
  _Float16* wWih12  = ws + WS_WIH12;
  _Float16* wWhh12  = ws + WS_WHH12;
  _Float16* wYs     = ws + WS_YS;

  int n;
  n = TSEQ * BSZ * DIN;
  transpose_x<<<(n + 255) / 256, 256, 0, stream>>>(x, wXT);
  n = 2 * G4 * DIN;
  f32_to_f16<<<(n + 255) / 256, 256, 0, stream>>>(Wih0f, wWih0, n);
  n = 2 * G4 * H;
  f32_to_f16<<<(n + 255) / 256, 256, 0, stream>>>(Whh0f, wWhh0, n);
  n = 4 * G4 * 2 * H;
  f32_to_f16<<<(n + 255) / 256, 256, 0, stream>>>(Wih12f, wWih12, n);
  n = 4 * G4 * H;
  f32_to_f16<<<(n + 255) / 256, 256, 0, stream>>>(Whh12f, wWhh12, n);

  // 48KB h + 96KB c + 64KB gates + 2KB x tile = 215040 B (< 320KB WGP LDS)
  const size_t shmem = (size_t)(6 * 16 * H * 2) + (size_t)(6 * 16 * H * 4)
                     + (size_t)(16 * G4 * 4) + (size_t)(16 * DIN * 2);
  lstm_recurrent<<<BSZ / 16, 512, shmem, stream>>>(wXT, wWih0, wWhh0, wWih12, wWhh12,
                                                   b0v, b12v, wYs);

  classify<<<TSEQ, 64, 0, stream>>>(wYs, convw, convb, (int*)d_out);
}
